// FasterMultiHeadAttention_30769145708882
// MI455X (gfx1250) — matrix operand order repaired and checked by reading, not compiled, not run on hardware
//
#include <hip/hip_runtime.h>

typedef __attribute__((ext_vector_type(16))) _Float16 v16h;
typedef __attribute__((ext_vector_type(8)))  _Float16 v8h;
typedef __attribute__((ext_vector_type(4)))  _Float16 v4h;
typedef __attribute__((ext_vector_type(8)))  float    v8f;
typedef __attribute__((ext_vector_type(4)))  float    v4f;

#define LQ   2048
#define LKV  2048
#define NH   16
#define DH   128
#define BM   64      // query rows per block (16 per wave, 4 waves)
#define BN   64      // kv rows per tile
#define NTHREADS 128

#define QK_STRIDE 136   // 128 + 8 halves pad: 16B-aligned rows, conflict-free frag loads
#define VT_STRIDE 72    // 64 + 8 halves pad
#define P_STRIDE  72

static __device__ __forceinline__ v16h pack16(v8h lo, v8h hi) {
  return __builtin_shufflevector(lo, hi, 0,1,2,3,4,5,6,7,8,9,10,11,12,13,14,15);
}

// raw v_exp_f32: base-2, flushes for very negative args (wanted for softmax tails)
static __device__ __forceinline__ float fast_exp2(float x) {
  return __builtin_amdgcn_exp2f(x);
}

__global__ __launch_bounds__(NTHREADS, 1)
void fa_wmma_kernel(const float* __restrict__ qg,
                    const float* __restrict__ kg,
                    const float* __restrict__ vg,
                    float* __restrict__ out)
{
  __shared__ __align__(16) _Float16 Qs[BM * QK_STRIDE];
  __shared__ __align__(16) _Float16 Ks[BN * QK_STRIDE];
  __shared__ __align__(16) _Float16 Vts[DH * VT_STRIDE];   // V transposed: [d][n]
  __shared__ __align__(16) _Float16 Ps[4 * 16 * P_STRIDE]; // per-wave P staging

  const int tid  = threadIdx.x;
  const int wave = tid >> 5;
  const int lane = tid & 31;
  const int half = lane >> 4;   // 0: lanes 0-15, 1: lanes 16-31
  const int l16  = lane & 15;

  const int q0 = blockIdx.x * BM;
  const int h  = blockIdx.y;

  // base-2 softmax domain: fold log2(e) into the 1/sqrt(d) score scale
  const float scale = 1.4426950408889634f * 0.08838834764831845f;

  // ---- load + convert Q tile (scaled) into LDS: float4 loads, packed-half stores ----
  for (int i = tid; i < (BM * DH) / 4; i += NTHREADS) {
    int r = i >> 5;            // (i*4) / 128
    int d = (i & 31) * 4;
    v4f x = *(const v4f*)(qg + (size_t)(q0 + r) * (NH * DH) + h * DH + d);
    v4h hx;
#pragma unroll
    for (int j = 0; j < 4; ++j) hx[j] = (_Float16)(x[j] * scale);
    *(v4h*)(Qs + r * QK_STRIDE + d) = hx;
  }
  __syncthreads();

  // ---- per-wave Q A-fragments (rows 16*wave..+15), 4 K-chunks of 32 ----
  v16h aq[4];
  {
    const _Float16* qrow = &Qs[(wave * 16 + l16) * QK_STRIDE];
#pragma unroll
    for (int c = 0; c < 4; ++c) {
      v8h lo = *(const v8h*)(qrow + c * 32 +      half * 8);
      v8h hi = *(const v8h*)(qrow + c * 32 + 16 + half * 8);
      aq[c] = pack16(lo, hi);
    }
  }

  // all-ones B matrix: rowsum(P) via the matrix pipe instead of lane butterflies
  v16h bone;
#pragma unroll
  for (int i = 0; i < 16; ++i) bone[i] = (_Float16)1.0f;

  const v8f vzero = {};
  float mrow[8], lrow[8], alpha[8];
#pragma unroll
  for (int r = 0; r < 8; ++r) { mrow[r] = -1e30f; lrow[r] = 0.f; }
  v8f oacc[8];
#pragma unroll
  for (int t = 0; t < 8; ++t) oacc[t] = vzero;

  for (int kv0 = 0; kv0 < LKV; kv0 += BN) {
    __syncthreads();
    // ---- load K tile (row-major) + V tile (transposed) into LDS, f32->f16 ----
    for (int i = tid; i < (BN * DH) / 4; i += NTHREADS) {
      int n = i >> 5;
      int d = (i & 31) * 4;
      size_t gidx = (size_t)(kv0 + n) * (NH * DH) + h * DH + d;
      v4f kf = *(const v4f*)(kg + gidx);
      v4f vf = *(const v4f*)(vg + gidx);
      v4h kh;
#pragma unroll
      for (int j = 0; j < 4; ++j) kh[j] = (_Float16)kf[j];
      *(v4h*)(Ks + n * QK_STRIDE + d) = kh;
#pragma unroll
      for (int j = 0; j < 4; ++j)
        Vts[(d + j) * VT_STRIDE + n] = (_Float16)vf[j];
    }
    // prefetch next tile into L2 while we compute this one
    if (kv0 + BN < LKV) {
      size_t pidx = (size_t)(kv0 + BN + (tid >> 1)) * (NH * DH) + h * DH + (tid & 1) * 64;
      __builtin_prefetch(&kg[pidx], 0, 1);
      __builtin_prefetch(&vg[pidx], 0, 1);
    }
    __syncthreads();

    // ---- S = (Q*scale) K^T : four 16x16 f32 tiles per wave ----
    v8f s[4];
#pragma unroll
    for (int j = 0; j < 4; ++j) {
      v8f c = vzero;
      const _Float16* krow = &Ks[(j * 16 + l16) * QK_STRIDE];
#pragma unroll
      for (int cc = 0; cc < 4; ++cc) {
        const _Float16* p = krow + cc * 32 + half * 8;    // B-frag: lane N, table-order K
        v16h b = pack16(*(const v8h*)p, *(const v8h*)(p + 16));
        c = __builtin_amdgcn_wmma_f32_16x16x32_f16(false, aq[cc], false, b,
                                                   (short)0, c, false, false);
      }
      s[j] = c;
    }

    // ---- online softmax (base-2); row (r + 8*half) spans one 16-lane group ----
#pragma unroll
    for (int r = 0; r < 8; ++r) {
      float mx = fmaxf(fmaxf(s[0][r], s[1][r]), fmaxf(s[2][r], s[3][r]));
#pragma unroll
      for (int d = 1; d < 16; d <<= 1) mx = fmaxf(mx, __shfl_xor(mx, d, 32));
      float mnew = fmaxf(mrow[r], mx);
      alpha[r]   = fast_exp2(mrow[r] - mnew);
      mrow[r]    = mnew;
#pragma unroll
      for (int j = 0; j < 4; ++j) s[j][r] = fast_exp2(s[j][r] - mnew);
#pragma unroll
      for (int t = 0; t < 8; ++t) oacc[t][r] *= alpha[r];
    }

    // ---- transpose P: C-layout -> A-layout via per-wave LDS staging ----
    _Float16* pw = &Ps[wave * 16 * P_STRIDE];
#pragma unroll
    for (int r = 0; r < 8; ++r) {
      int row = r + 8 * half;
#pragma unroll
      for (int j = 0; j < 4; ++j)
        pw[row * P_STRIDE + j * 16 + l16] = (_Float16)s[j][r];
    }
    asm volatile("s_wait_dscnt 0" ::: "memory");  // wave-private region: local wait suffices

    v16h ap[2];
    {
      const _Float16* prow = pw + l16 * P_STRIDE;
#pragma unroll
      for (int c = 0; c < 2; ++c) {
        v8h lo = *(const v8h*)(prow + c * 32 +      half * 8);
        v8h hi = *(const v8h*)(prow + c * 32 + 16 + half * 8);
        ap[c] = pack16(lo, hi);
      }
    }

    // ---- rowsum(P) on the matrix pipe: P x ones -> sum replicated per lane group ----
    {
      v8f rsum = __builtin_amdgcn_wmma_f32_16x16x32_f16(false, ap[0], false, bone,
                                                        (short)0, vzero, false, false);
      rsum     = __builtin_amdgcn_wmma_f32_16x16x32_f16(false, ap[1], false, bone,
                                                        (short)0, rsum, false, false);
#pragma unroll
      for (int r = 0; r < 8; ++r) lrow[r] = lrow[r] * alpha[r] + rsum[r];
    }

    // ---- O += P V : eight 16x16 f32 tiles per wave (Vt gives contiguous B-frags) ----
#pragma unroll
    for (int t = 0; t < 8; ++t) {
      v8f c = oacc[t];
      const _Float16* vrow = &Vts[(t * 16 + l16) * VT_STRIDE];
#pragma unroll
      for (int cc = 0; cc < 2; ++cc) {
        const _Float16* p = vrow + cc * 32 + half * 8;
        v16h b = pack16(*(const v8h*)p, *(const v8h*)(p + 16));
        c = __builtin_amdgcn_wmma_f32_16x16x32_f16(false, ap[cc], false, b,
                                                   (short)0, c, false, false);
      }
      oacc[t] = c;
    }
  }

  // ---- finalize: O /= l, write fp32 output [Lq, H, D] ----
#pragma unroll
  for (int r = 0; r < 8; ++r) {
    float inv = 1.f / lrow[r];
    int row = q0 + wave * 16 + r + 8 * half;
    float* orow = out + (size_t)row * (NH * DH) + h * DH;
#pragma unroll
    for (int t = 0; t < 8; ++t)
      orow[t * 16 + l16] = oacc[t][r] * inv;
  }
}

extern "C" void kernel_launch(void* const* d_in, const int* in_sizes, int n_in,
                              void* d_out, int out_size, void* d_ws, size_t ws_size,
                              hipStream_t stream) {
  (void)in_sizes; (void)n_in; (void)out_size; (void)d_ws; (void)ws_size;
  const float* q = (const float*)d_in[0];
  const float* k = (const float*)d_in[1];
  const float* v = (const float*)d_in[2];
  float* out = (float*)d_out;
  dim3 grid(LQ / BM, NH);
  fa_wmma_kernel<<<grid, NTHREADS, 0, stream>>>(q, k, v, out);
}
